// FastSpectralEncoder_17712445129493
// MI455X (gfx1250) — compile-verified, matrix operand order checked by reading, not run on hardware
//
#include <hip/hip_runtime.h>
#include <hip/hip_bf16.h>

typedef __attribute__((ext_vector_type(16))) _Float16 v16h;
typedef __attribute__((ext_vector_type(8)))  float    v8f;

#define KDIM 32
#define EPSN 1e-12f

// ---------------- degree / normalization ----------------

__global__ void deg_kernel(const int* __restrict__ ei, float* __restrict__ deg, int E) {
  int e = blockIdx.x * blockDim.x + threadIdx.x;
  if (e < E) atomicAdd(&deg[ei[e]], 1.0f);
}

__global__ void dis_kernel(const float* __restrict__ deg, float* __restrict__ dis, int n) {
  int i = blockIdx.x * blockDim.x + threadIdx.x;
  if (i < n) {
    float d = deg[i];
    dis[i] = (d > 0.0f) ? rsqrtf(fmaxf(d, EPSN)) : 0.0f;
  }
}

// per-column reduction over rows: MODE 0 -> sum A[i,c]^2, MODE 1 -> sum A[i,c]*B[i,c]
template <int MODE>
__global__ void colreduce_kernel(const float* __restrict__ A, const float* __restrict__ B,
                                 float* __restrict__ out32, int n) {
  __shared__ float s[256];
  const int c   = threadIdx.x & 31;
  const int g   = threadIdx.x >> 5;     // row-group within block (0..7)
  const int gpb = blockDim.x >> 5;      // 8 row-groups per block
  float local = 0.0f;
  for (int i = blockIdx.x * gpb + g; i < n; i += gridDim.x * gpb) {
    float a = A[i * KDIM + c];
    local += (MODE == 0) ? a * a : a * B[i * KDIM + c];
  }
  s[threadIdx.x] = local;
  __syncthreads();
  if (threadIdx.x < 32) {
    float sum = 0.0f;
    for (int gg = 0; gg < 8; ++gg) sum += s[gg * 32 + c];
    atomicAdd(&out32[c], sum);
  }
}

// dst[i,c] = src[i,c] / max(sqrt(normsq[c]), eps)
__global__ void scale_kernel(const float* __restrict__ src, const float* __restrict__ normsq,
                             float* __restrict__ dst, int n) {
  int t = blockIdx.x * blockDim.x + threadIdx.x;
  if (t < n * KDIM) {
    int c = t & 31;
    float nrm = sqrtf(normsq[c]);
    dst[t] = src[t] / fmaxf(nrm, EPSN);
  }
}

// ---------------- sparse A_norm @ V scatter ----------------
// one wave per edge, lane = k-column; Vout must be pre-zeroed.
__global__ void spmm_kernel(const int* __restrict__ ei, const float* __restrict__ dis,
                            const float* __restrict__ Vin, float* __restrict__ Vout, int E) {
  int t = blockIdx.x * blockDim.x + threadIdx.x;
  int e = t >> 5;
  int c = t & 31;
  if (e < E) {
    int s = ei[e];          // edge_index[0][e]
    int d = ei[E + e];      // edge_index[1][e]
    float w = dis[s] * dis[d];
    if (c == 0 && e + 8 < E) {
      int dn = ei[E + e + 8];
      __builtin_prefetch(&Vin[(size_t)dn * KDIM], 0, 0);   // global_prefetch_b8
    }
    atomicAdd(&Vout[(size_t)s * KDIM + c], w * Vin[(size_t)d * KDIM + c]);
  }
}

// lap = sort(1 - eigen) via rank-by-count (ties broken by index -> ranks are a permutation)
__global__ void lap_kernel(const float* __restrict__ eigen, float* __restrict__ lap) {
  int c = threadIdx.x;
  if (c < KDIM) {
    float v = 1.0f - eigen[c];
    int rank = 0;
    for (int j = 0; j < KDIM; ++j) {
      float u = 1.0f - eigen[j];
      rank += (u < v) || (u == v && j < c) ? 1 : 0;
    }
    lap[rank] = v;
  }
}

// ---------------- MLP tail via v_wmma_f32_16x16x32_f16 ----------------
// 16-bit A-matrix 16x32 layout: lane<16 holds row M=lane with K in {0..7,16..23};
// lane>=16 holds row M=lane-16 with K in {8..15,24..31}; half i -> VGPR i/2, pos i%2.
__device__ __forceinline__ int a_kidx(int lane, int i) {
  int v = i >> 1, p = i & 1;
  int base = (v < 4) ? (2 * v + p) : (16 + 2 * (v - 4) + p);
  return base + ((lane >= 16) ? 8 : 0);
}

__global__ __launch_bounds__(512) void mlp_kernel(const float* __restrict__ lap,
                                                  const float* __restrict__ W1,
                                                  const float* __restrict__ b1,
                                                  const float* __restrict__ W2,
                                                  const float* __restrict__ b2,
                                                  float* __restrict__ out) {
  __shared__ float h1s[256];
  const int lane = threadIdx.x & 31;
  const int wave = threadIdx.x >> 5;        // N-tile 0..15
  const int mrow = lane & 15;               // A-matrix row owned by this lane
  const int N    = wave * 16 + (lane & 15); // B/D column owned by this lane

  // layer 1: h1 = relu(lap @ W1 + b1), K=32 -> one WMMA per tile
  v16h a, b;
  for (int i = 0; i < 16; ++i) {
    int K = a_kidx(lane, i);
    a[i] = (mrow == 0) ? (_Float16)lap[K] : (_Float16)0.0f;
    b[i] = (_Float16)W1[K * 256 + N];
  }
  v8f c = {};
  c = __builtin_amdgcn_wmma_f32_16x16x32_f16(false, a, false, b, (short)0, c, false, false);
  if (lane < 16) h1s[N] = fmaxf(c[0] + b1[N], 0.0f);   // D row M=0 lives in vgpr0, lanes 0-15
  __syncthreads();

  // layer 2: out = h1 @ W2 + b2, K=256 -> 8 chained WMMAs per tile
  v8f acc = {};
  for (int kc = 0; kc < 8; ++kc) {
    v16h a2, bf;
    for (int i = 0; i < 16; ++i) {
      int K = kc * 32 + a_kidx(lane, i);
      a2[i] = (mrow == 0) ? (_Float16)h1s[K] : (_Float16)0.0f;
      bf[i] = (_Float16)W2[K * 256 + N];
    }
    acc = __builtin_amdgcn_wmma_f32_16x16x32_f16(false, a2, false, bf, (short)0, acc, false, false);
  }
  if (lane < 16) out[N] = acc[0] + b2[N];
}

// ---------------- driver ----------------

extern "C" void kernel_launch(void* const* d_in, const int* in_sizes, int n_in,
                              void* d_out, int out_size, void* d_ws, size_t ws_size,
                              hipStream_t stream) {
  (void)n_in; (void)out_size; (void)ws_size;
  const int*   ei    = (const int*)d_in[0];     // [2, E] row-major
  const float* Vinit = (const float*)d_in[2];   // [n, 32]
  const float* W1    = (const float*)d_in[3];   // [32, 256]
  const float* b1    = (const float*)d_in[4];   // [256]
  const float* W2    = (const float*)d_in[5];   // [256, 256]
  const float* b2    = (const float*)d_in[6];   // [256]
  const int E = in_sizes[0] / 2;
  const int n = in_sizes[2] / KDIM;

  // workspace layout (~3.2 MB total for n=8192)
  float* ws     = (float*)d_ws;
  float* deg    = ws;  ws += n;
  float* dis    = ws;  ws += n;
  float* VA     = ws;  ws += (size_t)n * KDIM;
  float* VB     = ws;  ws += (size_t)n * KDIM;
  float* AV     = ws;  ws += (size_t)n * KDIM;
  float* normsq = ws;  ws += KDIM;
  float* eigen  = ws;  ws += KDIM;
  float* lap    = ws;  ws += KDIM;

  const int TB = 256;
  auto nblk = [](long long t, int b) { return (int)((t + b - 1) / b); };

  // degrees -> D^-1/2
  hipMemsetAsync(deg, 0, sizeof(float) * (size_t)n, stream);
  deg_kernel<<<nblk(E, TB), TB, 0, stream>>>(ei, deg, E);
  dis_kernel<<<nblk(n, TB), TB, 0, stream>>>(deg, dis, n);

  // V = col_normalize(V_init)
  hipMemsetAsync(normsq, 0, sizeof(float) * KDIM, stream);
  colreduce_kernel<0><<<32, TB, 0, stream>>>(Vinit, nullptr, normsq, n);
  scale_kernel<<<nblk((long long)n * KDIM, TB), TB, 0, stream>>>(Vinit, normsq, VA, n);

  // 10 power iterations: VA = col_normalize(A_norm @ VA)
  for (int it = 0; it < 10; ++it) {
    hipMemsetAsync(VB, 0, sizeof(float) * (size_t)n * KDIM, stream);
    spmm_kernel<<<nblk((long long)E * 32, TB), TB, 0, stream>>>(ei, dis, VA, VB, E);
    hipMemsetAsync(normsq, 0, sizeof(float) * KDIM, stream);
    colreduce_kernel<0><<<32, TB, 0, stream>>>(VB, nullptr, normsq, n);
    scale_kernel<<<nblk((long long)n * KDIM, TB), TB, 0, stream>>>(VB, normsq, VA, n);
  }

  // AV = A_norm @ V ; eigen[c] = sum_i V[i,c]*AV[i,c]
  hipMemsetAsync(AV, 0, sizeof(float) * (size_t)n * KDIM, stream);
  spmm_kernel<<<nblk((long long)E * 32, TB), TB, 0, stream>>>(ei, dis, VA, AV, E);
  hipMemsetAsync(eigen, 0, sizeof(float) * KDIM, stream);
  colreduce_kernel<1><<<32, TB, 0, stream>>>(VA, AV, eigen, n);

  // lap = sort(1 - eigen); MLP tail on WMMA
  lap_kernel<<<1, 32, 0, stream>>>(eigen, lap);
  mlp_kernel<<<1, 512, 0, stream>>>(lap, W1, b1, W2, b2, (float*)d_out);
}
